// ModalityFusion_61529701482979
// MI455X (gfx1250) — matrix operand order repaired and checked by reading, not compiled, not run on hardware
//
#include <hip/hip_runtime.h>
#include <hip/hip_bf16.h>

// ---------------- types ----------------
typedef __attribute__((ext_vector_type(16))) __bf16 v16bf;
typedef __attribute__((ext_vector_type(8)))  float  v8f;
typedef __attribute__((ext_vector_type(4)))  unsigned int u32x4;
typedef __attribute__((ext_vector_type(2)))  unsigned int u32x2;
typedef __attribute__((ext_vector_type(4)))  float  f32x4;

union Frag { v16bf v; u32x4 u[2]; };

// ---------------- problem constants ----------------
static constexpr int L   = 64;
static constexpr int B   = 256;
static constexpr int D   = 1024;      // all three dims equal
static constexpr int LB  = L * B;     // 16384
static constexpr int K2  = 2048;      // concat dim

// workspace layout (bytes)
static constexpr size_t O_SEQBF  = 0;                                   // 3*LB*D bf16
static constexpr size_t O_FEATBF = O_SEQBF  + (size_t)3*LB*D*2;         // 3*B*D bf16
static constexpr size_t O_W1BF   = O_FEATBF + (size_t)3*B*D*2;          // 9*1024*2048 bf16
static constexpr size_t O_FPART  = O_W1BF   + (size_t)9*1024*2048*2;    // 9*B*D f32
static constexpr size_t O_S      = O_FPART  + (size_t)9*B*D*4;          // 9*LB f32
static constexpr size_t O_V      = O_S      + (size_t)9*LB*4;           // 3*B*D f32
static constexpr size_t O_T3     = O_V      + (size_t)3*B*D*4;          // 3*B*512 f32
static constexpr size_t O_S2     = O_T3     + (size_t)3*B*512*4;        // 3*B f32

struct PtrArr9  { const float* p[9]; };
struct MainPtrs { const float* w2[9]; const float* b2[9]; };
struct SeqPtrs  { const float* s[3]; };
struct HeadPtrs { const float* w1[3]; const float* b1[3];
                  const float* w2[3]; const float* b2[3];
                  const float* w3[3]; const float* b3[3]; };

// ---------------- fp32 -> bf16 conversion ----------------
__global__ __launch_bounds__(256) void cvt_f32_bf16(const float* __restrict__ src,
                                                    __bf16* __restrict__ dst, int n4) {
    int i = blockIdx.x * blockDim.x + threadIdx.x;
    int stride = gridDim.x * blockDim.x;
    for (; i < n4; i += stride) {
        f32x4 f = ((const f32x4*)src)[i];
        union { __bf16 h[4]; u32x2 u; } pk;
        pk.h[0] = (__bf16)f.x; pk.h[1] = (__bf16)f.y;
        pk.h[2] = (__bf16)f.z; pk.h[3] = (__bf16)f.w;
        ((u32x2*)dst)[i] = pk.u;
    }
}

// ---------------- fpart[mj][b][n] = f_sel @ Wb^T + b1  (WMMA bf16) ----------------
__global__ __launch_bounds__(256) void fpart_kernel(const __bf16* __restrict__ featbf,
                                                    const __bf16* __restrict__ w1bf,
                                                    float* __restrict__ fpart,
                                                    PtrArr9 b1ptrs) {
    int lane = threadIdx.x & 31;
    int wave = threadIdx.x >> 5;
    int w    = blockIdx.x * 8 + wave;          // 0..9215
    int mj   = w >> 10;                        // 0..8
    int rem  = w & 1023;
    int mt   = rem >> 6;                       // 16 M-tiles (B=256 rows)
    int nt   = rem & 63;                       // 64 N-tiles (D=1024 cols)
    int m    = mj / 3, j = mj - m * 3;
    int fsel = (j == 0) ? m : (j == 1 ? (m == 0 ? 1 : 0) : (m == 2 ? 1 : 2));
    int ln   = lane & 15;
    int kh   = lane >> 4;

    const __bf16* aPtr = featbf + ((size_t)fsel * B + mt * 16 + ln) * D;
    const __bf16* bPtr = w1bf + ((size_t)mj * 1024 + nt * 16 + ln) * K2 + 1024; // Wb half

    v8f c;
    #pragma unroll
    for (int i = 0; i < 8; ++i) c[i] = 0.0f;

    for (int kk = 0; kk < 1024; kk += 32) {
        Frag a, b;
        a.u[0] = *(const u32x4*)(aPtr + kk + kh * 8);
        a.u[1] = *(const u32x4*)(aPtr + kk + kh * 8 + 16);
        b.u[0] = *(const u32x4*)(bPtr + kk + kh * 8);
        b.u[1] = *(const u32x4*)(bPtr + kk + kh * 8 + 16);
        c = __builtin_amdgcn_wmma_f32_16x16x32_bf16(false, a.v, false, b.v,
                                                    (short)0, c, false, false);
    }
    int col = nt * 16 + ln;
    float bias = b1ptrs.p[mj][col];
    #pragma unroll
    for (int r = 0; r < 8; ++r) {
        int row = mt * 16 + r + kh * 8;
        fpart[((size_t)mj * B + row) * D + col] = c[r] + bias;
    }
}

// ---------------- fused main GEMM: s = (tanh(seq@Wa^T + fpart)) @ W2^T + b2 ----------------
// 2x2 register blocking: each wave owns 2 M-tiles x 2 N-tiles -> 4 WMMAs per K-step
// fed by 4 global + 4 LDS loads (2 mem ops per WMMA).
__global__ __launch_bounds__(256) void fused_gemm_kernel(const __bf16* __restrict__ seqbf,
                                                         const __bf16* __restrict__ w1bf,
                                                         const float* __restrict__ fpart,
                                                         float* __restrict__ sout,
                                                         MainPtrs P) {
    extern __shared__ __align__(16) char smem[];
    __bf16* Bs = (__bf16*)smem;                          // 32 rows x 1024 K bf16 = 64 KB

    int lane  = threadIdx.x & 31;
    int wave  = threadIdx.x >> 5;
    int mj    = blockIdx.x >> 6;                         // 0..8
    int strip = blockIdx.x & 63;                         // 64 strips of 256 rows (one l each)
    int m     = mj / 3;
    int rbase = strip * 256 + wave * 32;                 // wave: rows [rbase, rbase+32)
    int ln    = lane & 15;
    int kh    = lane >> 4;

    const __bf16* aPtr0 = seqbf + ((size_t)m * LB + rbase + ln) * D;
    const __bf16* aPtr1 = aPtr0 + (size_t)16 * D;
    const __bf16* wrow  = w1bf + (size_t)mj * 1024 * K2; // Wa = first 1024 cols of each row
    const float*  w2p   = P.w2[mj];
    float         b2v   = P.b2[mj][0];
    const float*  fpb   = fpart + (size_t)mj * B * D;

    int tnn  = threadIdx.x >> 3;                         // 0..31 : weight row within tile
    int tseg = threadIdx.x & 7;                          // 0..7  : 256B segment within row

    float sacc0[8], sacc1[8];
    #pragma unroll
    for (int r = 0; r < 8; ++r) { sacc0[r] = 0.0f; sacc1[r] = 0.0f; }

    for (int n0 = 0; n0 < 1024; n0 += 32) {
        __syncthreads();
        {   // cooperative stage of B tile: weight rows n0..n0+31, K=0..1023 (2KB/row)
            const u32x4* src = (const u32x4*)(wrow + (size_t)(n0 + tnn) * K2);
            u32x4* dst = (u32x4*)Bs + tnn * 128;
            #pragma unroll
            for (int i = 0; i < 16; ++i) dst[tseg * 16 + i] = src[tseg * 16 + i];
        }
        __syncthreads();
        if (n0 + 32 < 1024)  // pull next weight tile toward caches (global_prefetch_b8)
            __builtin_prefetch(wrow + (size_t)(n0 + 32 + tnn) * K2 + tseg * 256, 0, 0);

        v8f c00, c01, c10, c11;
        #pragma unroll
        for (int i = 0; i < 8; ++i) { c00[i] = 0.0f; c01[i] = 0.0f; c10[i] = 0.0f; c11[i] = 0.0f; }

        const u32x4* bls0 = (const u32x4*)Bs + (((ln)      * 1024 + kh * 8) >> 3);
        const u32x4* bls1 = (const u32x4*)Bs + (((ln + 16) * 1024 + kh * 8) >> 3);
        #pragma unroll 2
        for (int kk = 0; kk < 1024; kk += 32) {
            Frag a0, a1, b0, b1;
            a0.u[0] = *(const u32x4*)(aPtr0 + kk + kh * 8);
            a0.u[1] = *(const u32x4*)(aPtr0 + kk + kh * 8 + 16);
            a1.u[0] = *(const u32x4*)(aPtr1 + kk + kh * 8);
            a1.u[1] = *(const u32x4*)(aPtr1 + kk + kh * 8 + 16);
            b0.u[0] = bls0[kk >> 3];
            b0.u[1] = bls0[(kk >> 3) + 2];
            b1.u[0] = bls1[kk >> 3];
            b1.u[1] = bls1[(kk >> 3) + 2];
            c00 = __builtin_amdgcn_wmma_f32_16x16x32_bf16(false, a0.v, false, b0.v,
                                                          (short)0, c00, false, false);
            c01 = __builtin_amdgcn_wmma_f32_16x16x32_bf16(false, a0.v, false, b1.v,
                                                          (short)0, c01, false, false);
            c10 = __builtin_amdgcn_wmma_f32_16x16x32_bf16(false, a1.v, false, b0.v,
                                                          (short)0, c10, false, false);
            c11 = __builtin_amdgcn_wmma_f32_16x16x32_bf16(false, a1.v, false, b1.v,
                                                          (short)0, c11, false, false);
        }
        // fused epilogue: tanh + dot with W2 row, keep partial scores in registers
        float w2v0 = w2p[n0 + ln];
        float w2v1 = w2p[n0 + 16 + ln];
        #pragma unroll
        for (int r = 0; r < 8; ++r) {
            int row0 = rbase + r + kh * 8;
            int row1 = row0 + 16;
            int b0i  = row0 & (B - 1);
            int b1i  = row1 & (B - 1);
            const float* f0 = fpb + (size_t)b0i * D + n0 + ln;
            const float* f1 = fpb + (size_t)b1i * D + n0 + ln;
            sacc0[r] += tanhf(c00[r] + f0[0])  * w2v0 + tanhf(c01[r] + f0[16]) * w2v1;
            sacc1[r] += tanhf(c10[r] + f1[0])  * w2v0 + tanhf(c11[r] + f1[16]) * w2v1;
        }
    }
    // reduce over the 16 lanes holding the N dimension
    #pragma unroll
    for (int r = 0; r < 8; ++r) {
        float v0 = sacc0[r];
        v0 += __shfl_xor(v0, 1, 32);
        v0 += __shfl_xor(v0, 2, 32);
        v0 += __shfl_xor(v0, 4, 32);
        v0 += __shfl_xor(v0, 8, 32);
        float v1 = sacc1[r];
        v1 += __shfl_xor(v1, 1, 32);
        v1 += __shfl_xor(v1, 2, 32);
        v1 += __shfl_xor(v1, 4, 32);
        v1 += __shfl_xor(v1, 8, 32);
        if (ln == 0) {
            int row = rbase + r + kh * 8;
            sout[(size_t)mj * LB + row]      = v0 + b2v;
            sout[(size_t)mj * LB + row + 16] = v1 + b2v;
        }
    }
}

// ---------------- softmax over L + weighted mean of seq ----------------
__global__ __launch_bounds__(128) void fusion_kernel(const float* __restrict__ sbuf,
                                                     SeqPtrs seqs,
                                                     float* __restrict__ vout) {
    int m = blockIdx.x >> 8;
    int b = blockIdx.x & 255;
    __shared__ float sv[3][L];
    __shared__ float wl[L];
    int t = threadIdx.x;
    if (t < L) {
        for (int j = 0; j < 3; ++j)
            sv[j][t] = sbuf[(size_t)(m * 3 + j) * LB + t * B + b];
    }
    __syncthreads();
    if (t < L) {
        float acc = 0.0f;
        for (int j = 0; j < 3; ++j) {
            float mx = -1e30f;
            for (int l = 0; l < L; ++l) mx = fmaxf(mx, sv[j][l]);
            float den = 0.0f;
            for (int l = 0; l < L; ++l) den += __expf(sv[j][l] - mx);
            acc += __expf(sv[j][t] - mx) / den;
        }
        wl[t] = acc;
    }
    __syncthreads();
    const float* seq = seqs.s[m];
    for (int d = t; d < D; d += 128) {
        float acc = 0.0f;
        for (int l = 0; l < L; ++l)
            acc += wl[l] * seq[((size_t)l * B + b) * D + d];
        vout[((size_t)m * B + b) * D + d] = acc * (1.0f / (3.0f * (float)L));
    }
}

// ---------------- small heads: s2 scalar score + t3 projection ----------------
__global__ __launch_bounds__(256) void head_kernel(const float* __restrict__ vout,
                                                   HeadPtrs P,
                                                   float* __restrict__ t3,
                                                   float* __restrict__ s2) {
    int m = blockIdx.x >> 8;
    int b = blockIdx.x & 255;
    int n1 = (m == 0) ? 250 : 512;
    const float* v = vout + ((size_t)m * B + b) * D;
    __shared__ float red[256];
    int t = threadIdx.x;

    float part = 0.0f;
    const float* W1 = P.w1[m]; const float* B1 = P.b1[m]; const float* W2 = P.w2[m];
    for (int f = t; f < n1; f += 256) {
        const float* wr = W1 + (size_t)f * D;
        float acc = B1[f];
        for (int k = 0; k < D; ++k) acc += v[k] * wr[k];
        part += tanhf(acc) * W2[f];
    }
    red[t] = part;
    __syncthreads();
    for (int s = 128; s > 0; s >>= 1) {
        if (t < s) red[t] += red[t + s];
        __syncthreads();
    }
    if (t == 0) s2[m * B + b] = red[0] + P.b2[m][0];

    const float* W3 = P.w3[m]; const float* B3 = P.b3[m];
    for (int f = t; f < 512; f += 256) {
        const float* wr = W3 + (size_t)f * D;
        float acc = B3[f];
        for (int k = 0; k < D; ++k) acc += v[k] * wr[k];
        t3[((size_t)m * B + b) * 512 + f] = tanhf(acc);
    }
}

// ---------------- final softmax-over-modality mix ----------------
__global__ __launch_bounds__(512) void final_kernel(const float* __restrict__ s2,
                                                    const float* __restrict__ t3,
                                                    float* __restrict__ out) {
    int b = blockIdx.x;
    int f = threadIdx.x;
    float a0 = s2[b], a1 = s2[B + b], a2 = s2[2 * B + b];
    float mx = fmaxf(a0, fmaxf(a1, a2));
    float e0 = __expf(a0 - mx), e1 = __expf(a1 - mx), e2 = __expf(a2 - mx);
    float inv = 1.0f / (e0 + e1 + e2);
    float r = e0 * inv * t3[((size_t)0 * B + b) * 512 + f]
            + e1 * inv * t3[((size_t)1 * B + b) * 512 + f]
            + e2 * inv * t3[((size_t)2 * B + b) * 512 + f];
    out[(size_t)b * 512 + f] = r;
}

// ---------------- host driver ----------------
extern "C" void kernel_launch(void* const* d_in, const int* in_sizes, int n_in,
                              void* d_out, int out_size, void* d_ws, size_t ws_size,
                              hipStream_t stream) {
    (void)in_sizes; (void)n_in; (void)out_size; (void)ws_size;
    char* ws = (char*)d_ws;
    __bf16* seqbf  = (__bf16*)(ws + O_SEQBF);
    __bf16* featbf = (__bf16*)(ws + O_FEATBF);
    __bf16* w1bf   = (__bf16*)(ws + O_W1BF);
    float*  fpart  = (float*) (ws + O_FPART);
    float*  sbuf   = (float*) (ws + O_S);
    float*  vout   = (float*) (ws + O_V);
    float*  t3     = (float*) (ws + O_T3);
    float*  s2     = (float*) (ws + O_S2);

    // --- convert seqs / features / W1 weights to bf16 ---
    for (int m = 0; m < 3; ++m) {
        // seq inputs at indices 1,3,5 ; feature inputs at 0,2,4
        cvt_f32_bf16<<<8192, 256, 0, stream>>>((const float*)d_in[1 + 2 * m],
                                               seqbf + (size_t)m * LB * D, (LB * D) / 4);
        cvt_f32_bf16<<<256, 256, 0, stream>>>((const float*)d_in[0 + 2 * m],
                                              featbf + (size_t)m * B * D, (B * D) / 4);
        for (int j = 0; j < 3; ++j) {
            int mj = m * 3 + j;
            cvt_f32_bf16<<<2048, 256, 0, stream>>>((const float*)d_in[6 + m * 12 + j],
                                                   w1bf + (size_t)mj * 1024 * K2,
                                                   (1024 * K2) / 4);
        }
    }

    // --- attention param pointer tables ---
    PtrArr9 b1p; MainPtrs mp;
    for (int m = 0; m < 3; ++m)
        for (int j = 0; j < 3; ++j) {
            int mj = m * 3 + j;
            b1p.p[mj]  = (const float*)d_in[6 + m * 12 + 3 + j];
            mp.w2[mj]  = (const float*)d_in[6 + m * 12 + 6 + j];
            mp.b2[mj]  = (const float*)d_in[6 + m * 12 + 9 + j];
        }

    // --- feature-side partial (L-invariant) via WMMA ---
    fpart_kernel<<<1152, 256, 0, stream>>>(featbf, w1bf, fpart, b1p);

    // --- fused seq GEMM + tanh + W2 GEMV -> attention scores (64 KB dynamic LDS) ---
    fused_gemm_kernel<<<9 * 64, 256, 65536, stream>>>(seqbf, w1bf, fpart, sbuf, mp);

    // --- softmax over L + weighted mean of seq ---
    SeqPtrs sp;
    sp.s[0] = (const float*)d_in[1];
    sp.s[1] = (const float*)d_in[3];
    sp.s[2] = (const float*)d_in[5];
    fusion_kernel<<<3 * 256, 128, 0, stream>>>(sbuf, sp, vout);

    // --- heads ---
    HeadPtrs hp;
    for (int m = 0; m < 3; ++m) {
        hp.w1[m] = (const float*)d_in[42 + 2 * m];
        hp.b1[m] = (const float*)d_in[43 + 2 * m];
        hp.w2[m] = (const float*)d_in[48 + 2 * m];
        hp.b2[m] = (const float*)d_in[49 + 2 * m];
        hp.w3[m] = (const float*)d_in[54 + 2 * m];
        hp.b3[m] = (const float*)d_in[55 + 2 * m];
    }
    head_kernel<<<3 * 256, 256, 0, stream>>>(vout, hp, t3, s2);

    // --- final softmax mix ---
    final_kernel<<<B, 512, 0, stream>>>(s2, t3, (float*)d_out);
}
